// GraphAttention_27358941675733
// MI455X (gfx1250) — compile-verified, matrix operand order checked by reading, not run on hardware
//
#include <hip/hip_runtime.h>
#include <math.h>

// ---------------------------------------------------------------------------
// GAT layer for MI455X (gfx1250, wave32).
// All GEMMs via v_wmma_f32_16x16x32_bf16 (bf16 A/B, f32 accum).
// Weights pre-packed into concatenated uniform-stride bf16 buffers (branch-
// free wave-uniform B addressing); activation tiles staged in LDS as bf16;
// B fragments batch-loaded per k-step so loads clause together and WMMAs
// overlap outstanding loads instead of waiting on loadcnt==0 each tile.
// ---------------------------------------------------------------------------

typedef __bf16 bf16_t;
typedef __attribute__((ext_vector_type(16))) __bf16 v16bf;
typedef __attribute__((ext_vector_type(8)))  __bf16 v8bf;
typedef __attribute__((ext_vector_type(4)))  __bf16 v4bf;
typedef __attribute__((ext_vector_type(8)))  float  v8f;

#define NODE_DIM  128
#define EDGE_DIM  128
#define HDIM      256   // NUM_HEADS * HIDDEN_DIM
#define NHEADS    8
#define HIDDEN    32
#define HEAD_DIM  16
#define NTAB      640   // 256 (ni_h+bni) + 256 (nj_h+bnj) + 128 (nj@Wm_nj.T)
#define NEG_SLOPE 0.2f

// Load a 16x32 bf16 fragment (A-style) from a row-major bf16 matrix.
// Lane L (<16): row r0+L, K in {k0+0..7, k0+16..23}; lane L(>=16): row r0+L-16,
// K in {k0+8..15, k0+24..31}.  (CDNA5 ISA 7.12.2 16-bit A layout.)
// Applying the same loader to row-major W[n][k] yields the B fragment for
// C = X @ W.T  (lane = output column n, same K striping).
__device__ __forceinline__ v16bf load_frag_bf16(const bf16_t* M, int ld, int r0, int k0) {
  const int lane = threadIdx.x & 31;
  const int r    = r0 + (lane & 15);
  const int half = lane >> 4;
  const bf16_t* base = M + (long)r * ld + k0 + half * 8;
  v8bf lo = *reinterpret_cast<const v8bf*>(base);
  v8bf hi = *reinterpret_cast<const v8bf*>(base + 16);
  return __builtin_shufflevector(lo, hi, 0, 1, 2, 3, 4, 5, 6, 7,
                                 8, 9, 10, 11, 12, 13, 14, 15);
}

__device__ __forceinline__ v8f wmma_bf16(v16bf a, v16bf b, v8f c) {
  return __builtin_amdgcn_wmma_f32_16x16x32_bf16(false, a, false, b, (short)0, c,
                                                 false, false);
}

// Cooperative stage of a row-major f32 tile into LDS as bf16.
__device__ __forceinline__ void stage_tile_bf16(const float* __restrict__ g,
                                                bf16_t* s, int nElem,
                                                int tid, int nthr) {
  const float4* gs = reinterpret_cast<const float4*>(g);
  for (int i = tid; i < nElem / 4; i += nthr) {
    float4 v = gs[i];
    v4bf p;
    p[0] = (bf16_t)v.x; p[1] = (bf16_t)v.y;
    p[2] = (bf16_t)v.z; p[3] = (bf16_t)v.w;
    *reinterpret_cast<v4bf*>(s + i * 4) = p;
  }
}

// order-preserving float -> uint map for atomicMax segment-max
__device__ __forceinline__ unsigned f2u_mono(float f) {
  int b = __float_as_int(f);
  return (b >= 0) ? ((unsigned)b | 0x80000000u) : ~(unsigned)b;
}
__device__ __forceinline__ float u2f_mono(unsigned u) {
  int b = (u & 0x80000000u) ? (int)(u & 0x7fffffffu) : (int)~u;
  return __int_as_float(b);
}

// ---------------------------------------------------------------------------
// K0: zero the accumulator region (seg_max_u, seg_sum, agg) each call.
__global__ void zero_kernel(float* p, long n) {
  long i = (long)blockIdx.x * 256 + threadIdx.x;
  if (i < n) p[i] = 0.0f;
}

// Kpack: build concatenated bf16 weight buffers + bias_cat (tiny, once/launch).
//   wnode_cat[640][128] = Wni(256x128) ++ Wnj(256x128) ++ Wm[:, :128](128x128)
//   wedge_cat[384][128] = We(256x128)  ++ Wm[:, 128:](128x128)
//   wo_bf[128][128]     = Wo
//   bias_cat[640]       = bni ++ bnj ++ 0
__global__ void pack_node_w_kernel(const float* __restrict__ Wni,
                                   const float* __restrict__ Wnj,
                                   const float* __restrict__ Wm,
                                   bf16_t* __restrict__ dst) {
  int i = blockIdx.x * 256 + threadIdx.x;           // 640*128
  if (i >= NTAB * NODE_DIM) return;
  int r = i >> 7, k = i & 127;
  float v;
  if (r < 256)      v = Wni[r * NODE_DIM + k];
  else if (r < 512) v = Wnj[(r - 256) * NODE_DIM + k];
  else              v = Wm[(r - 512) * (NODE_DIM + EDGE_DIM) + k];
  dst[i] = (bf16_t)v;
}
__global__ void pack_edge_w_kernel(const float* __restrict__ We,
                                   const float* __restrict__ Wm,
                                   bf16_t* __restrict__ dst) {
  int i = blockIdx.x * 256 + threadIdx.x;           // 384*128
  if (i >= 384 * EDGE_DIM) return;
  int r = i >> 7, k = i & 127;
  float v;
  if (r < 256) v = We[r * EDGE_DIM + k];
  else         v = Wm[(r - 256) * (NODE_DIM + EDGE_DIM) + NODE_DIM + k];
  dst[i] = (bf16_t)v;
}
__global__ void pack_wo_bias_kernel(const float* __restrict__ Wo,
                                    const float* __restrict__ bni,
                                    const float* __restrict__ bnj,
                                    bf16_t* __restrict__ wo_bf,
                                    float* __restrict__ bias_cat) {
  int i = blockIdx.x * 256 + threadIdx.x;
  if (i < NODE_DIM * NODE_DIM) wo_bf[i] = (bf16_t)Wo[i];
  if (i < NTAB) {
    float b = 0.0f;
    if (i < 256)      b = bni[i];
    else if (i < 512) b = bnj[i - 256];
    bias_cat[i] = b;
  }
}

// ---------------------------------------------------------------------------
// K1: per-node GEMM  X[N,128] @ wnode_cat^T -> node_tab[N,640]
// One 16-row tile per block; unit = wave + 8*u (branch-free addressing).
// B fragments batch-loaded per k-step (independent regs -> clause + overlap).
__global__ __launch_bounds__(256) void node_proj_kernel(
    const float* __restrict__ nodef, const bf16_t* __restrict__ wnode_cat,
    const float* __restrict__ bias_cat, float* __restrict__ node_tab, int N) {
  __shared__ __align__(16) bf16_t sA[16 * NODE_DIM];
  const int rt  = blockIdx.x;
  const int tid = threadIdx.x;
  stage_tile_bf16(nodef + (long)rt * 16 * NODE_DIM, sA, 16 * NODE_DIM, tid, 256);
  __syncthreads();

  const int wave = tid >> 5;
  const int lane = tid & 31;

  v8f acc[5] = {};
#pragma unroll
  for (int kk = 0; kk < 4; ++kk) {
    v16bf a = load_frag_bf16(sA, NODE_DIM, 0, kk * 32);
    v16bf b[5];
#pragma unroll
    for (int u = 0; u < 5; ++u)
      b[u] = load_frag_bf16(wnode_cat, NODE_DIM, (wave + 8 * u) * 16, kk * 32);
#pragma unroll
    for (int u = 0; u < 5; ++u)
      acc[u] = wmma_bf16(a, b[u], acc[u]);
  }

  const int c = lane & 15;
  const int hrow = lane >> 4;
#pragma unroll
  for (int u = 0; u < 5; ++u) {
    const int n = (wave + 8 * u) * 16 + c;   // column in node_tab [0,640)
    const float bias = bias_cat[n];
#pragma unroll
    for (int i = 0; i < 8; ++i) {
      int m = rt * 16 + i + 8 * hrow;
      node_tab[(long)m * NTAB + n] = acc[u][i] + bias;
    }
  }
}

// ---------------------------------------------------------------------------
// K2: fused per-edge kernel.  16-edge tile per block (256 thr = 8 waves).
// unit = wave + 8*u:  u=0,1 -> hidden col-tiles 0..15 (attn logits path);
//                     u=2   -> message col-tiles 16..23.
__global__ __launch_bounds__(256) void edge_kernel(
    const float* __restrict__ edge_feat, const int* __restrict__ edge_index,
    const float* __restrict__ node_tab, const bf16_t* __restrict__ wedge_cat,
    const float* __restrict__ be, const float* __restrict__ bm,
    const float* __restrict__ attn_proj,
    float* __restrict__ logits, float* __restrict__ messages,
    unsigned* __restrict__ seg_max_u, int E) {
  __shared__ __align__(16) bf16_t sA[16 * EDGE_DIM];
  __shared__ int   sSrc[16];
  __shared__ int   sDst[16];
  __shared__ float sLogit[16 * NHEADS];

  const int e0  = blockIdx.x * 16;
  const int tid = threadIdx.x;

  stage_tile_bf16(edge_feat + (long)e0 * EDGE_DIM, sA, 16 * EDGE_DIM, tid, 256);
  if (tid < 16) {
    sSrc[tid] = edge_index[e0 + tid];      // aggregation index i
    sDst[tid] = edge_index[E + e0 + tid];  // neighbor j
  }
  if (tid < 16 * NHEADS) sLogit[tid] = 0.0f;
  __syncthreads();

  const int wave = tid >> 5;
  const int lane = tid & 31;

  v8f acc[3] = {};
#pragma unroll
  for (int kk = 0; kk < 4; ++kk) {
    v16bf a = load_frag_bf16(sA, EDGE_DIM, 0, kk * 32);
    v16bf b[3];
#pragma unroll
    for (int u = 0; u < 3; ++u)
      b[u] = load_frag_bf16(wedge_cat, EDGE_DIM, (wave + 8 * u) * 16, kk * 32);
#pragma unroll
    for (int u = 0; u < 3; ++u)
      acc[u] = wmma_bf16(a, b[u], acc[u]);
  }

  const int c    = lane & 15;
  const int hrow = lane >> 4;
#pragma unroll
  for (int u = 0; u < 3; ++u) {
    if (u < 2) {                              // compile-time branch
      const int n    = (wave + 8 * u) * 16 + c;   // hidden column [0,256)
      const int head = n >> 5;
      const float ap = attn_proj[(n >> 5) * HIDDEN + (n & 31)];
      const float bb = be[n];
#pragma unroll
      for (int i = 0; i < 8; ++i) {
        const int m = i + 8 * hrow;
        float v = acc[u][i]
                + node_tab[(long)sSrc[m] * NTAB + n]
                + node_tab[(long)sDst[m] * NTAB + 256 + n]
                + bb;
        v = (v >= 0.0f) ? v : NEG_SLOPE * v;           // leaky relu
        atomicAdd(&sLogit[m * NHEADS + head], v * ap); // ds_add_f32
      }
    } else {
      const int n  = wave * 16 + c;           // message column [0,128)
      const float bb = bm[n];
#pragma unroll
      for (int i = 0; i < 8; ++i) {
        const int m = i + 8 * hrow;
        float v = acc[u][i] + node_tab[(long)sDst[m] * NTAB + 512 + n] + bb;
        messages[(long)(e0 + m) * NODE_DIM + n] = v;
      }
    }
  }
  __syncthreads();

  if (tid < 16 * NHEADS) {
    const int m = tid >> 3, h = tid & 7;
    const float lg = sLogit[tid];
    logits[(long)(e0 + m) * NHEADS + h] = lg;
    atomicMax(&seg_max_u[(long)sSrc[m] * NHEADS + h], f2u_mono(lg));
  }
}

// ---------------------------------------------------------------------------
// K3: ex = exp(logit - seg_max[src]); seg_sum[src] += ex  (overwrites logits)
__global__ void softmax_pass_kernel(const int* __restrict__ edge_index,
                                    const unsigned* __restrict__ seg_max_u,
                                    float* __restrict__ logits,
                                    float* __restrict__ seg_sum, int E) {
  long idx = (long)blockIdx.x * 256 + threadIdx.x;
  if (idx >= (long)E * NHEADS) return;
  const int e = (int)(idx >> 3), h = (int)(idx & 7);
  const int s = edge_index[e];
  const float mx = u2f_mono(seg_max_u[(long)s * NHEADS + h]);
  const float ex = __expf(logits[idx] - mx);
  logits[idx] = ex;
  atomicAdd(&seg_sum[(long)s * NHEADS + h], ex);
}

// ---------------------------------------------------------------------------
// K4: agg[src] += messages[e] * (ex / seg_sum[src]) per head (f32 atomics)
__global__ void scatter_kernel(const int* __restrict__ edge_index,
                               const float* __restrict__ ex,
                               const float* __restrict__ seg_sum,
                               const float* __restrict__ messages,
                               float* __restrict__ agg, int E) {
  long idx = (long)blockIdx.x * 256 + threadIdx.x;   // E * 32 threads
  if (idx >= (long)E * 32) return;
  const int e  = (int)(idx >> 5);
  const int g  = (int)(idx & 31);
  const int n0 = g * 4;               // 4 contiguous cols, same head (HEAD_DIM=16)
  const int head = n0 >> 4;
  const int s  = edge_index[e];
  const float sc = ex[(long)e * NHEADS + head] / seg_sum[(long)s * NHEADS + head];
  float4 m = *reinterpret_cast<const float4*>(messages + (long)e * NODE_DIM + n0);
  float* a = agg + (long)s * NODE_DIM + n0;
  atomicAdd(a + 0, m.x * sc);
  atomicAdd(a + 1, m.y * sc);
  atomicAdd(a + 2, m.z * sc);
  atomicAdd(a + 3, m.w * sc);
}

// ---------------------------------------------------------------------------
// K5: out = agg @ Wo^T + bo   (N x 128 x 128); one 16-row tile per block.
__global__ __launch_bounds__(256) void out_proj_kernel(
    const float* __restrict__ agg, const bf16_t* __restrict__ wo_bf,
    const float* __restrict__ bo, float* __restrict__ out, int N) {
  __shared__ __align__(16) bf16_t sA[16 * NODE_DIM];
  const int rt  = blockIdx.x;
  const int tid = threadIdx.x;
  stage_tile_bf16(agg + (long)rt * 16 * NODE_DIM, sA, 16 * NODE_DIM, tid, 256);
  __syncthreads();

  const int wave = tid >> 5;   // 8 waves = 8 col tiles
  const int lane = tid & 31;

  v8f acc = {};
#pragma unroll
  for (int kk = 0; kk < 4; ++kk) {
    v16bf a = load_frag_bf16(sA, NODE_DIM, 0, kk * 32);
    v16bf b = load_frag_bf16(wo_bf, NODE_DIM, wave * 16, kk * 32);
    acc = wmma_bf16(a, b, acc);
  }
  const int c = lane & 15;
  const int hrow = lane >> 4;
  const int n = wave * 16 + c;
  const float bias = bo[n];
#pragma unroll
  for (int i = 0; i < 8; ++i) {
    int m = rt * 16 + i + 8 * hrow;
    out[(long)m * NODE_DIM + n] = acc[i] + bias;
  }
}

// ---------------------------------------------------------------------------
extern "C" void kernel_launch(void* const* d_in, const int* in_sizes, int n_in,
                              void* d_out, int out_size, void* d_ws, size_t ws_size,
                              hipStream_t stream) {
  (void)n_in; (void)out_size; (void)ws_size;
  const float* node_feat = (const float*)d_in[0];
  const float* edge_feat = (const float*)d_in[1];
  const int*   edge_idx  = (const int*)d_in[2];
  const float* Wni = (const float*)d_in[3];
  const float* bni = (const float*)d_in[4];
  const float* Wnj = (const float*)d_in[5];
  const float* bnj = (const float*)d_in[6];
  const float* We  = (const float*)d_in[7];
  const float* be  = (const float*)d_in[8];
  const float* ap  = (const float*)d_in[9];
  const float* Wm  = (const float*)d_in[10];
  const float* bm  = (const float*)d_in[11];
  const float* Wo  = (const float*)d_in[12];
  const float* bo  = (const float*)d_in[13];
  float* out = (float*)d_out;

  const int N = in_sizes[0] / NODE_DIM;   // 20000 (multiple of 16)
  const int E = in_sizes[1] / EDGE_DIM;   // 320000 (multiple of 16)

  // Workspace layout: f32 region, then packed bf16 weights, then bias_cat
  float*    ws        = (float*)d_ws;
  float*    node_tab  = ws;                                   // N*640
  float*    logits    = node_tab + (size_t)N * NTAB;          // E*8 (later: ex)
  float*    messages  = logits + (size_t)E * NHEADS;          // E*128
  unsigned* seg_max_u = (unsigned*)(messages + (size_t)E * NODE_DIM);  // N*8
  float*    seg_sum   = (float*)(seg_max_u + (size_t)N * NHEADS);      // N*8
  float*    agg       = seg_sum + (size_t)N * NHEADS;                  // N*128
  bf16_t*   wnode_cat = (bf16_t*)(agg + (size_t)N * NODE_DIM);         // 640*128
  bf16_t*   wedge_cat = wnode_cat + NTAB * NODE_DIM;                   // 384*128
  bf16_t*   wo_bf     = wedge_cat + 384 * EDGE_DIM;                    // 128*128
  float*    bias_cat  = (float*)(wo_bf + NODE_DIM * NODE_DIM);         // 640

  // K0: zero seg_max_u + seg_sum + agg (contiguous, N*144 words)
  {
    long nz = (long)N * (NHEADS + NHEADS + NODE_DIM);
    zero_kernel<<<(int)((nz + 255) / 256), 256, 0, stream>>>((float*)seg_max_u, nz);
  }
  // Kpack: build packed bf16 weights + bias_cat (L2-resident afterwards)
  pack_node_w_kernel<<<(NTAB * NODE_DIM + 255) / 256, 256, 0, stream>>>(Wni, Wnj, Wm, wnode_cat);
  pack_edge_w_kernel<<<(384 * EDGE_DIM + 255) / 256, 256, 0, stream>>>(We, Wm, wedge_cat);
  pack_wo_bias_kernel<<<(NODE_DIM * NODE_DIM + 255) / 256, 256, 0, stream>>>(Wo, bni, bnj, wo_bf, bias_cat);

  // K1: node projections
  node_proj_kernel<<<N / 16, 256, 0, stream>>>(node_feat, wnode_cat, bias_cat,
                                               node_tab, N);
  // K2: fused edge GEMMs + logits + seg-max
  edge_kernel<<<E / 16, 256, 0, stream>>>(edge_feat, edge_idx, node_tab,
                                          wedge_cat, be, bm, ap, logits,
                                          messages, seg_max_u, E);
  // K3: softmax normalize pass
  {
    long n = (long)E * NHEADS;
    softmax_pass_kernel<<<(int)((n + 255) / 256), 256, 0, stream>>>(
        edge_idx, seg_max_u, logits, seg_sum, E);
  }
  // K4: weighted scatter-add of messages
  {
    long n = (long)E * 32;
    scatter_kernel<<<(int)((n + 255) / 256), 256, 0, stream>>>(
        edge_idx, logits, seg_sum, messages, agg, E);
  }
  // K5: output projection
  out_proj_kernel<<<N / 16, 256, 0, stream>>>(agg, wo_bf, bo, out, N);
}
